// PCNNMILAttention_23003844838100
// MI455X (gfx1250) — matrix operand order repaired and checked by reading, not compiled, not run on hardware
//
#include <hip/hip_runtime.h>
#include <float.h>
#include <math.h>

typedef __attribute__((ext_vector_type(16))) _Float16 v16h;
typedef __attribute__((ext_vector_type(8)))  _Float16 v8h;
typedef __attribute__((ext_vector_type(4)))  _Float16 v4h;
typedef __attribute__((ext_vector_type(8)))  float    v8f;

#define L_SEQ  512
#define EMB_D  128
#define NFILT  128
#define HID_D  384
#define ATT_D  128
#define NCLS_D 53
#define N_INST 512
#define ROWP   136            // padded halfs per LDS row (272 B: 16B-aligned, conflict-free)
#define NROWS  (L_SEQ + 2)    // +2 zero halo rows for 'same' padding

// ---------------------------------------------------------------------------
// Kernel 1: embedding gather -> LDS(f16), conv-as-GEMM via WMMA, fused
//           bias+ReLU+piecewise max pool. One block per instance.
// ---------------------------------------------------------------------------
extern "C" __global__ __launch_bounds__(256)
void pcnn_conv_pool(const int*   __restrict__ char_ids,
                    const int*   __restrict__ pos_e1,
                    const int*   __restrict__ pos_e2,
                    const float* __restrict__ emb,
                    const float* __restrict__ conv_w,   // [F][E][K] f32
                    const float* __restrict__ conv_b,
                    float*       __restrict__ H)        // [N][3*F]
{
    extern __shared__ _Float16 xs[];   // NROWS * ROWP halfs (~137 KB)
    const int inst = blockIdx.x;
    const int tid  = threadIdx.x;
    const int wave = tid >> 5;
    const int lane = tid & 31;

    // ---- Phase 1: gather embedding rows to LDS as f16 (zero halo rows) ----
    for (int r = wave; r < NROWS; r += 8) {
        v4h h4 = {};
        if (r != 0 && r != NROWS - 1) {
            const int l   = r - 1;
            const int cid = char_ids[inst * L_SEQ + l];
            const float4 f4 = *(const float4*)(emb + (size_t)cid * EMB_D + lane * 4);
            h4[0] = (_Float16)f4.x; h4[1] = (_Float16)f4.y;
            h4[2] = (_Float16)f4.z; h4[3] = (_Float16)f4.w;
        }
        *(v4h*)(xs + (size_t)r * ROWP + lane * 4) = h4;
    }
    __syncthreads();

    // ---- Phase 2: preload this wave's 12 B fragments (f32 -> f16) ----
    // Wave w owns filters f = w*16 .. w*16+15.  B fragment (32x16 KxN):
    // lane holds column n = lane&15; lanes<16 hold K rows 0..7,16..23 of the
    // 32-chunk, lanes>=16 hold K rows 8..15,24..31 (two halfs per VGPR).
    const int ncol = lane & 15;
    const int f    = wave * 16 + ncol;
    const int ehi  = (lane < 16) ? 0 : 16;

    v16h Bf[12];
    #pragma unroll
    for (int kk = 0; kk < 3; ++kk) {
        #pragma unroll
        for (int kc = 0; kc < 4; ++kc) {
            v16h b;
            #pragma unroll
            for (int j = 0; j < 16; ++j) {
                const int e = kc * 32 + (ehi >> 1) + (j & 7) + ((j >> 3) << 4);
                b[j] = (_Float16)conv_w[((size_t)f * EMB_D + e) * 3 + kk];
            }
            Bf[kk * 4 + kc] = b;
        }
    }

    // ---- Segment boundaries (mirrors reference clamping logic) ----
    const int p1 = pos_e1[inst], p2 = pos_e2[inst];
    int e1 = (p1 < p2) ? p1 : p2;
    int e2 = (p1 < p2) ? p2 : p1;
    e1 = max(0, min(e1, L_SEQ));
    e2 = max(0, min(e2, L_SEQ));
    if (e1 == e2) e2 = min(e1 + 1, L_SEQ);
    const int end1 = (e1 > 0) ? e1 : 1;

    const float bias = conv_b[f];
    float mx1 = -FLT_MAX, mx2 = -FLT_MAX, mx3 = -FLT_MAX;

    const int rowoff    = lane & 15;            // A-fragment row m = lane&15
    const int ab0       = (lane < 16) ? 0 : 8;  // A K-offset per ISA layout
    const int mbase_add = (lane < 16) ? 0 : 8;  // C-tile row offset per ISA layout

    // ---- Phase 3: 32 row tiles x (3 shifts x 4 K-chunks) WMMAs + pooling ----
    for (int mt = 0; mt < 32; ++mt) {
        v8f c = {};
        #pragma unroll
        for (int kk = 0; kk < 3; ++kk) {
            // l + (kk-1) in halo coords = l + kk ; halo guarantees in-range
            const _Float16* rp = xs + (size_t)(mt * 16 + rowoff + kk) * ROWP;
            #pragma unroll
            for (int kc = 0; kc < 4; ++kc) {
                const v8h lo = *(const v8h*)(rp + kc * 32 + ab0);        // K 0..7 / 8..15
                const v8h hi = *(const v8h*)(rp + kc * 32 + ab0 + 16);   // K 16..23 / 24..31
                v16h a;
                #pragma unroll
                for (int j = 0; j < 8; ++j) { a[j] = lo[j]; a[j + 8] = hi[j]; }
                c = __builtin_amdgcn_wmma_f32_16x16x32_f16(
                        false, a, false, Bf[kk * 4 + kc], (short)0, c, false, false);
            }
        }
        const int lbase = mt * 16 + mbase_add;
        #pragma unroll
        for (int r = 0; r < 8; ++r) {
            const int   l   = lbase + r;
            const float val = fmaxf(c[r] + bias, 0.0f);
            if (l < end1)            mx1 = fmaxf(mx1, val);
            if (l >= e1 && l < e2)   mx2 = fmaxf(mx2, val);
            const bool in3 = (e2 < L_SEQ) ? (l >= e2) : (l == L_SEQ - 1);
            if (in3)                 mx3 = fmaxf(mx3, val);
        }
    }

    // fold the two half-lanes that share a filter column (wave32 shuffle)
    mx1 = fmaxf(mx1, __shfl_xor(mx1, 16, 32));
    mx2 = fmaxf(mx2, __shfl_xor(mx2, 16, 32));
    mx3 = fmaxf(mx3, __shfl_xor(mx3, 16, 32));

    if (lane < 16) {
        float* hrow = H + (size_t)inst * HID_D;
        hrow[f]             = mx1;
        hrow[NFILT + f]     = mx2;
        hrow[2 * NFILT + f] = mx3;
    }
}

// ---------------------------------------------------------------------------
// Kernel 2: scores[n] = tanh(H[n] @ W^T + b) . u   (one block per instance)
// ---------------------------------------------------------------------------
extern "C" __global__ __launch_bounds__(128)
void att_scores(const float* __restrict__ H,
                const float* __restrict__ W_w,   // [ATT][HID]
                const float* __restrict__ W_b,
                const float* __restrict__ u_w,
                float*       __restrict__ scores)
{
    __shared__ float part[ATT_D];
    const int n = blockIdx.x;
    const int t = threadIdx.x;
    const float* h = H   + (size_t)n * HID_D;
    const float* w = W_w + (size_t)t * HID_D;
    float acc = W_b[t];
    #pragma unroll 4
    for (int j = 0; j < HID_D; ++j) acc = fmaf(w[j], h[j], acc);
    part[t] = tanhf(acc) * u_w[t];
    __syncthreads();
    for (int s = 64; s > 0; s >>= 1) {
        if (t < s) part[t] += part[t + s];
        __syncthreads();
    }
    if (t == 0) scores[n] = part[0];
}

// ---------------------------------------------------------------------------
// Kernel 3 (single block): softmax over N, h_bag = attn @ H, logits.
// out = [logits(53) | attn(512)]
// ---------------------------------------------------------------------------
extern "C" __global__ __launch_bounds__(512)
void finalize(const float* __restrict__ H,
              const float* __restrict__ scores,
              const float* __restrict__ fc_w,   // [NCLS][HID]
              const float* __restrict__ fc_b,
              float*       __restrict__ out)
{
    __shared__ float attn_s[N_INST];
    __shared__ float red[N_INST];
    __shared__ float hbag[HID_D];
    const int t = threadIdx.x;

    const float sc = scores[t];
    red[t] = sc;
    __syncthreads();
    for (int s = 256; s > 0; s >>= 1) {
        if (t < s) red[t] = fmaxf(red[t], red[t + s]);
        __syncthreads();
    }
    const float mx = red[0];
    __syncthreads();

    const float ex = __expf(sc - mx);
    red[t] = ex;
    __syncthreads();
    for (int s = 256; s > 0; s >>= 1) {
        if (t < s) red[t] += red[t + s];
        __syncthreads();
    }
    const float at = ex / red[0];
    attn_s[t] = at;
    out[NCLS_D + t] = at;
    __syncthreads();

    if (t < HID_D) {
        float acc = 0.0f;
        for (int n2 = 0; n2 < N_INST; ++n2)
            acc = fmaf(attn_s[n2], H[(size_t)n2 * HID_D + t], acc);
        hbag[t] = acc;
    }
    __syncthreads();

    if (t < NCLS_D) {
        float acc = fc_b[t];
        const float* w = fc_w + (size_t)t * HID_D;
        for (int j = 0; j < HID_D; ++j) acc = fmaf(w[j], hbag[j], acc);
        out[t] = acc;
    }
}

// ---------------------------------------------------------------------------
extern "C" void kernel_launch(void* const* d_in, const int* in_sizes, int n_in,
                              void* d_out, int out_size, void* d_ws, size_t ws_size,
                              hipStream_t stream)
{
    const int*   char_ids = (const int*)  d_in[0];
    const int*   pos_e1   = (const int*)  d_in[1];
    const int*   pos_e2   = (const int*)  d_in[2];
    const float* emb      = (const float*)d_in[3];
    const float* conv_w   = (const float*)d_in[4];
    const float* conv_b   = (const float*)d_in[5];
    const float* W_w      = (const float*)d_in[6];
    const float* W_b      = (const float*)d_in[7];
    const float* u_w      = (const float*)d_in[8];
    const float* fc_w     = (const float*)d_in[9];
    const float* fc_b     = (const float*)d_in[10];

    float* out    = (float*)d_out;
    float* H      = (float*)d_ws;                 // 512*384 f32
    float* scores = H + (size_t)N_INST * HID_D;   // 512 f32

    const size_t smem = (size_t)NROWS * ROWP * sizeof(_Float16);  // ~137 KB

    pcnn_conv_pool<<<N_INST, 256, smem, stream>>>(char_ids, pos_e1, pos_e2,
                                                  emb, conv_w, conv_b, H);
    att_scores<<<N_INST, 128, 0, stream>>>(H, W_w, W_b, u_w, scores);
    finalize<<<1, 512, 0, stream>>>(H, scores, fc_w, fc_b, out);
}
